// MotionAttentionModel_16200616640926
// MI455X (gfx1250) — matrix operand order repaired and checked by reading, not compiled, not run on hardware
//
#include <hip/hip_runtime.h>
#include <hip/hip_bf16.h>

// ---------------- types ----------------
typedef __attribute__((ext_vector_type(16))) __bf16 v16bf;
typedef __attribute__((ext_vector_type(8)))  __bf16 v8bf;
typedef __attribute__((ext_vector_type(8)))  float  v8f;
typedef __attribute__((ext_vector_type(4)))  unsigned int u32x4;
typedef __attribute__((ext_vector_type(4)))  int i32x4;
typedef __attribute__((ext_vector_type(8)))  int i32x8;

#define WMMA_BF16(a,b,c) \
  __builtin_amdgcn_wmma_f32_16x16x32_bf16(false,(a),false,(b),(short)0,(c),false,false)

// ---------------- problem constants ----------------
static const int kB  = 64;   // batch
static const int kN  = 64;   // history length
static const int kJ  = 15;   // joints
static const int kD  = 9;    // joint dim
static const int kJD = 135;  // kJ*kD
static const int kK  = 40;   // DCT coeffs (padded to 64 for layer-1 K)
static const int kT  = 24;   // future steps
static const int kIT = 9;    // ceil(135/16) i-tiles per batch
static const int kNB = 4;    // history steps batched per iteration (B-frag reuse x4)

// adjacency neighbor lists (A = eye + bones, both directions), hardcoded
__device__ __constant__ int c_nb_off[16] = {0,4,9,12,14,17,19,22,24,27,30,32,35,38,40,41};
__device__ __constant__ int c_nb[41] = {
  0,1,8,11,  0,1,2,4,6,  1,2,3,  2,3,  1,4,5,  4,5,  1,6,7,  6,7,
  0,8,9,  8,9,10,  9,10,  0,11,12,  11,12,13,  12,13,  14
};

// ---------------- WMMA fragment helpers ----------------
// A-matrix 16x32 bf16 per-wave layout (ISA 7.12.2):
//  lanes 0-15 : row M=lane,   elems 0..7 -> K=0..7,  elems 8..15 -> K=16..23
//  lanes16-31 : row M=lane-16,elems 0..7 -> K=8..15, elems 8..15 -> K=24..31
__device__ inline v16bf load_a_frag(const __bf16* base, int ld, int lane) {
  int m    = lane & 15;
  int kofs = (lane >> 4) << 3;     // 0 or 8
  const __bf16* p = base + m * ld + kofs;
  v8bf lo = *(const v8bf*)(p);
  v8bf hi = *(const v8bf*)(p + 16);
  v16bf a;
#pragma unroll
  for (int e = 0; e < 8; ++e) { a[e] = lo[e]; a[e + 8] = hi[e]; }
  return a;
}

// B-matrix 32x16 bf16: pre-packed so each lane loads 16 contiguous bf16 (32B).
__device__ inline v16bf load_b_frag(const __bf16* packed, int lane) {
  return *(const v16bf*)(packed + lane * 16);
}

// ---------------- TDM: async 3D tile load (K=40 f32) x (16 i rows) x (nb n steps) ----------------
// D# per CDNA5 ISA 8.x: group0 {count|lds_addr|global_addr|type=2},
// group1 {data_size=4B, tensor_dim0=40, tensor_dim1=rows_avail (HW zero-fills OOB i rows),
//         tile 40x16xnb, dim0 stride 40, dim1 stride 5400=135*40}, group2 {tensor_dim2=nb}.
__device__ inline void tdm_load_hist_tile(const float* gptr, unsigned lds_off, int rows_avail) {
  unsigned long long ga = (unsigned long long)(size_t)gptr;
  u32x4 g0 = { 1u,                                   // count=1 (valid), user mode
               lds_off,                              // LDS byte address
               (unsigned)(ga & 0xffffffffull),
               (unsigned)((ga >> 32) & 0x1ffffffull) | (2u << 30) };  // type=2 "image"
  i32x8 g1 = { (int)(2u << 16),                      // data_size = 4 bytes
               (int)(40u << 16),                     // tensor_dim0 = 40 (bits 63:48)
               (int)(((unsigned)rows_avail & 0xffffu) << 16),  // tensor_dim1 lo16 (bits 95:80)
               (int)(40u << 16),                     // tile_dim0 = 40 (bits 127:112)
               (int)(16u | ((unsigned)kNB << 16)),   // tile_dim1=16, tile_dim2=kNB
               40,                                   // tensor_dim0_stride = 40
               (int)(5400u << 16),                   // tensor_dim1_stride lo16 (=135*40)
               0 };
  i32x4 g2 = { kNB, 0, 0, 0 };                       // tensor_dim2 = kNB
  i32x4 g3 = { 0, 0, 0, 0 };
  i32x8 g4 = { 0, 0, 0, 0, 0, 0, 0, 0 };             // unused (6-arg toolchain form)
  __builtin_amdgcn_tensor_load_to_lds(g0, g1, g2, g3, g4, 0);
}

// ---------------- weight pack kernel ----------------
__global__ void pack_weight(const float* __restrict__ W, __bf16* __restrict__ out,
                            int K, int Ncols, int Kp, int Np) {
  int idx = blockIdx.x * blockDim.x + threadIdx.x;
  int total = Kp * Np;
  if (idx >= total) return;
  int ntiles = Np >> 4;
  int e    = idx & 15;
  int lane = (idx >> 4) & 31;
  int frag = idx >> 9;
  int nt = frag % ntiles;
  int kt = frag / ntiles;
  int n = nt * 16 + (lane & 15);
  int k = kt * 32 + ((lane >> 4) << 3) + (e & 7) + ((e >> 3) << 4);
  float v = (k < K && n < Ncols) ? W[(size_t)k * Ncols + n] : 0.0f;
  out[idx] = (__bf16)v;
}

// ---------------- 16-row fused 2-layer MLP tile (used by Q encoder) ----------------
__device__ inline void mlp_tile(const __bf16* X,        // LDS [16][64]
                                const __bf16* W1p, const float* b1,
                                const __bf16* W2p, const float* b2,
                                __bf16* Hl, float* Ol, int wave, int lane) {
  v8f acc1[2];
#pragma unroll
  for (int s = 0; s < 2; ++s) {
    int nt = wave + s * 8;
    float bv = b1[nt * 16 + (lane & 15)];
    v8f acc;
#pragma unroll
    for (int r = 0; r < 8; ++r) acc[r] = bv;
#pragma unroll
    for (int kt = 0; kt < 2; ++kt) {
      v16bf a  = load_a_frag(X + kt * 32, 64, lane);
      v16bf bb = load_b_frag(W1p + (size_t)(kt * 16 + nt) * 512, lane);
      acc = WMMA_BF16(a, bb, acc);
    }
    acc1[s] = acc;
  }
#pragma unroll
  for (int s = 0; s < 2; ++s) {
    int nt = wave + s * 8;
    int n  = nt * 16 + (lane & 15);
    int m0 = (lane >> 4) << 3;
#pragma unroll
    for (int r = 0; r < 8; ++r) {
      float v = acc1[s][r];
      v = v > 0.0f ? v : 0.0f;
      Hl[(m0 + r) * 256 + n] = (__bf16)v;
    }
  }
  __syncthreads();
#pragma unroll
  for (int s = 0; s < 2; ++s) {
    int nt = wave + s * 8;
    float bv = b2[nt * 16 + (lane & 15)];
    v8f acc;
#pragma unroll
    for (int r = 0; r < 8; ++r) acc[r] = bv;
#pragma unroll
    for (int kt = 0; kt < 8; ++kt) {
      v16bf a  = load_a_frag(Hl + kt * 32, 256, lane);
      v16bf bb = load_b_frag(W2p + (size_t)(kt * 16 + nt) * 512, lane);
      acc = WMMA_BF16(a, bb, acc);
    }
    int n  = nt * 16 + (lane & 15);
    int m0 = (lane >> 4) << 3;
#pragma unroll
    for (int r = 0; r < 8; ++r) Ol[(m0 + r) * 256 + n] = acc[r];
  }
  __syncthreads();
}

// ---------------- 64-row fused 2-layer MLP (4 row tiles; B fragment reused x4) ----------------
__device__ inline void mlp_tile4(const __bf16* X,   // LDS [4][16][64]
                                 const __bf16* W1p, const float* b1,
                                 const __bf16* W2p, const float* b2,
                                 __bf16* Hl,        // LDS [64][256] bf16
                                 __bf16* Olb,       // LDS [64][256] bf16
                                 int wave, int lane) {
#pragma unroll
  for (int s = 0; s < 2; ++s) {
    int nt = wave + s * 8;
    float bv = b1[nt * 16 + (lane & 15)];
    v8f acc[4];
#pragma unroll
    for (int rt = 0; rt < 4; ++rt)
#pragma unroll
      for (int r = 0; r < 8; ++r) acc[rt][r] = bv;
#pragma unroll
    for (int kt = 0; kt < 2; ++kt) {
      v16bf bb = load_b_frag(W1p + (size_t)(kt * 16 + nt) * 512, lane);
#pragma unroll
      for (int rt = 0; rt < 4; ++rt) {
        v16bf a = load_a_frag(X + rt * 1024 + kt * 32, 64, lane);
        acc[rt] = WMMA_BF16(a, bb, acc[rt]);
      }
    }
    int n  = nt * 16 + (lane & 15);
    int m0 = (lane >> 4) << 3;
#pragma unroll
    for (int rt = 0; rt < 4; ++rt)
#pragma unroll
      for (int r = 0; r < 8; ++r) {
        float v = acc[rt][r];
        v = v > 0.0f ? v : 0.0f;
        Hl[(rt * 16 + m0 + r) * 256 + n] = (__bf16)v;
      }
  }
  __syncthreads();
#pragma unroll
  for (int s = 0; s < 2; ++s) {
    int nt = wave + s * 8;
    float bv = b2[nt * 16 + (lane & 15)];
    v8f acc[4];
#pragma unroll
    for (int rt = 0; rt < 4; ++rt)
#pragma unroll
      for (int r = 0; r < 8; ++r) acc[rt][r] = bv;
#pragma unroll
    for (int kt = 0; kt < 8; ++kt) {
      v16bf bb = load_b_frag(W2p + (size_t)(kt * 16 + nt) * 512, lane);
#pragma unroll
      for (int rt = 0; rt < 4; ++rt) {
        v16bf a = load_a_frag(Hl + rt * 16 * 256 + kt * 32, 256, lane);
        acc[rt] = WMMA_BF16(a, bb, acc[rt]);
      }
    }
    int n  = nt * 16 + (lane & 15);
    int m0 = (lane >> 4) << 3;
#pragma unroll
    for (int rt = 0; rt < 4; ++rt)
#pragma unroll
      for (int r = 0; r < 8; ++r)
        Olb[(rt * 16 + m0 + r) * 256 + n] = (__bf16)acc[rt][r];
  }
  __syncthreads();
}

// ---------------- kernel 1: Q = MLP(x_flat) ----------------
__global__ __launch_bounds__(256)
void qmlp_kernel(const float* __restrict__ x,
                 const __bf16* __restrict__ qw1p, const float* __restrict__ qb1,
                 const __bf16* __restrict__ qw2p, const float* __restrict__ qb2,
                 float* __restrict__ Qws) {
  __shared__ __bf16 X[16 * 64];
  __shared__ __bf16 Hl[16 * 256];
  __shared__ float  Ol[16 * 256];
  int b = blockIdx.x / kIT, it = blockIdx.x % kIT;
  int t = threadIdx.x, wave = t >> 5, lane = t & 31;
  for (int idx = t; idx < 16 * 64; idx += 256) {
    int r = idx >> 6, k = idx & 63;
    int i = it * 16 + r;
    float v = (k < kK && i < kJD) ? x[((size_t)b * kJD + i) * kK + k] : 0.0f;
    X[idx] = (__bf16)v;
  }
  __syncthreads();
  mlp_tile(X, qw1p, qb1, qw2p, qb2, Hl, Ol, wave, lane);
  float* dst = Qws + ((size_t)b * 144 + it * 16) * 256;
  for (int idx = t; idx < 16 * 256; idx += 256) dst[idx] = Ol[idx];
}

// ---------------- kernel 2: fused K/V MLP + attention, TDM-staged, NB=4 ----------------
__global__ __launch_bounds__(256)
void attn_kernel(const float* __restrict__ hist, const float* __restrict__ Qws,
                 const __bf16* __restrict__ kw1p, const float* __restrict__ kb1,
                 const __bf16* __restrict__ kw2p, const float* __restrict__ kb2,
                 const __bf16* __restrict__ vw1p, const float* __restrict__ vb1,
                 const __bf16* __restrict__ vw2p, const float* __restrict__ vb2,
                 __bf16* __restrict__ features) {
  __shared__ float  Xf[2][kNB * 16 * kK];   // TDM staging, double-buffered (f32)
  __shared__ __bf16 X[kNB * 16 * 64];       // bf16, K padded 40->64
  __shared__ __bf16 Hl[64 * 256];
  __shared__ __bf16 Olb[64 * 256];
  __shared__ __bf16 Qbb[16 * 256];
  __shared__ float  Sc[16 * 64];
  __shared__ float  Cx[16 * 256];
  __shared__ float  Pp[64 * 16];

  int b = blockIdx.x / kIT, it = blockIdx.x % kIT;
  int t = threadIdx.x, wave = t >> 5, lane = t & 31;
  int rows_avail = kJD - it * 16;           // HW zero-fills i rows beyond this

  const float* qsrc = Qws + ((size_t)b * 144 + it * 16) * 256;
  for (int idx = t; idx < 16 * 256; idx += 256) {
    Qbb[idx] = (__bf16)qsrc[idx];
    Cx[idx]  = 0.0f;
  }
  __syncthreads();

  for (int pass = 0; pass < 2; ++pass) {
    const __bf16* W1p = pass ? vw1p : kw1p;
    const float*  B1  = pass ? vb1  : kb1;
    const __bf16* W2p = pass ? vw2p : kw2p;
    const float*  B2  = pass ? vb2  : kb2;

    // prologue: async-load group 0 into buffer 0
    if (wave == 0)
      tdm_load_hist_tile(hist + (((size_t)b * kN + 0) * kJD + it * 16) * kK,
                         (unsigned)(size_t)&Xf[0][0], rows_avail);

    for (int ng = 0; ng < kN / kNB; ++ng) {
      int buf = ng & 1;
      if (wave == 0) {
        if (ng < kN / kNB - 1) {
          tdm_load_hist_tile(hist + (((size_t)b * kN + (ng + 1) * kNB) * kJD + it * 16) * kK,
                             (unsigned)(size_t)&Xf[buf ^ 1][0], rows_avail);
          __builtin_amdgcn_s_wait_tensorcnt(1);   // group ng complete
        } else {
          __builtin_amdgcn_s_wait_tensorcnt(0);
        }
      }
      __syncthreads();

      // convert staged f32 -> bf16, pad K 40->64
      for (int idx = t; idx < kNB * 16 * 64; idx += 256) {
        int k = idx & 63, row = idx >> 6;
        float v = (k < kK) ? Xf[buf][row * kK + k] : 0.0f;
        X[idx] = (__bf16)v;
      }
      __syncthreads();

      mlp_tile4(X, W1p, B1, W2p, B2, Hl, Olb, wave, lane);

      if (pass == 0) {
        // scores: row-wise dot(Q, K) for 4 n's
        int i = t >> 4, seg = t & 15;
#pragma unroll
        for (int ns = 0; ns < kNB; ++ns) {
          float p = 0.0f;
#pragma unroll
          for (int e = 0; e < 16; ++e) {
            int c = seg * 16 + e;
            p += (float)Qbb[i * 256 + c] * (float)Olb[(ns * 16 + i) * 256 + c];
          }
          Pp[(ns * 16 + i) * 16 + seg] = p;
        }
        __syncthreads();
        if (t < 64) {
          int ns = t >> 4, i2 = t & 15;
          float s = 0.0f;
#pragma unroll
          for (int g = 0; g < 16; ++g) s += Pp[(ns * 16 + i2) * 16 + g];
          Sc[i2 * 64 + ng * kNB + ns] = s * 0.0625f;   // 1/sqrt(256)
        }
        __syncthreads();
      } else {
        // context: Cx += attn * V for 4 n's
        int i = t >> 4, c0 = (t & 15) * 16;
#pragma unroll
        for (int ns = 0; ns < kNB; ++ns) {
          float a = Sc[i * 64 + ng * kNB + ns];
#pragma unroll
          for (int e = 0; e < 16; ++e)
            Cx[i * 256 + c0 + e] += a * (float)Olb[(ns * 16 + i) * 256 + c0 + e];
        }
        __syncthreads();
      }
    }

    if (pass == 0) {
      // softmax over n (one thread per i-row)
      if (t < 16) {
        float mx = -1e30f;
        for (int n = 0; n < kN; ++n) mx = fmaxf(mx, Sc[t * 64 + n]);
        float sum = 0.0f;
        for (int n = 0; n < kN; ++n) {
          float e = __expf(Sc[t * 64 + n] - mx);
          Sc[t * 64 + n] = e;
          sum += e;
        }
        float inv = 1.0f / sum;
        for (int n = 0; n < kN; ++n) Sc[t * 64 + n] *= inv;
      }
      __syncthreads();
    }
  }

  // features = concat(Q, ctx) bf16, layout [b][j][d*512 + {0..255 | 256..511}]
  for (int idx = t; idx < 16 * 256; idx += 256) {
    int r = idx >> 8, c = idx & 255;
    int i = it * 16 + r;
    if (i < kJD) {
      int j = i / kD, d = i % kD;
      __bf16* f = features + ((size_t)b * kJ + j) * 4608 + d * 512;
      f[c]       = Qbb[idx];
      f[256 + c] = (__bf16)Cx[idx];
    }
  }
}

// ---------------- generic bf16 WMMA GEMM: Out[R][Np] = A[R][Kp] @ Bpacked ----------------
__global__ __launch_bounds__(256)
void gemm_kernel(const __bf16* __restrict__ A, const __bf16* __restrict__ Bp,
                 float* __restrict__ Out, int Kp, int Nvalid, int Np) {
  __shared__ __bf16 As[16 * 256];
  int ntiles  = Np >> 4;
  int cgroups = (ntiles + 7) >> 3;
  int rt = blockIdx.x / cgroups, cg = blockIdx.x % cgroups;
  int t = threadIdx.x, wave = t >> 5, lane = t & 31;
  int nt = cg * 8 + wave;
  v8f acc;
#pragma unroll
  for (int r = 0; r < 8; ++r) acc[r] = 0.0f;

  int kouter = Kp >> 8;
  int r0 = t >> 4, c0 = (t & 15) * 16;
  for (int ko = 0; ko < kouter; ++ko) {
    int kbase = ko * 256;
    *(v16bf*)&As[r0 * 256 + c0] =
        *(const v16bf*)&A[((size_t)(rt * 16 + r0)) * Kp + kbase + c0];
    __syncthreads();
    if (nt < ntiles) {
#pragma unroll
      for (int kt = 0; kt < 8; ++kt) {
        v16bf a  = load_a_frag(As + kt * 32, 256, lane);
        v16bf bb = load_b_frag(Bp + (size_t)((ko * 8 + kt) * ntiles + nt) * 512, lane);
        acc = WMMA_BF16(a, bb, acc);
      }
    }
    __syncthreads();
  }
  if (nt < ntiles) {
    int n = nt * 16 + (lane & 15);
    if (n < Nvalid) {
      int m0 = (lane >> 4) << 3;
#pragma unroll
      for (int r = 0; r < 8; ++r)
        Out[((size_t)(rt * 16 + m0 + r)) * Np + n] = acc[r];
    }
  }
}

// ---------------- graph conv stages ----------------
__global__ void graph1_kernel(const float* __restrict__ Y, const float* __restrict__ g1b,
                              __bf16* __restrict__ x1) {
  int idx = blockIdx.x * blockDim.x + threadIdx.x;
  if (idx >= kB * kJ * 256) return;
  int c = idx & 255;
  int row = idx >> 8;
  int b = row / kJ, j = row % kJ;
  float s = g1b[c];
  for (int e = c_nb_off[j]; e < c_nb_off[j + 1]; ++e)
    s += Y[((size_t)b * kJ + c_nb[e]) * 256 + c];
  s = s > 0.0f ? s : 0.0f;
  x1[idx] = (__bf16)s;
}

__global__ void graph2_kernel(const float* __restrict__ Wo, const float* __restrict__ g2b,
                              float* __restrict__ x2) {
  int idx = blockIdx.x * blockDim.x + threadIdx.x;
  if (idx >= kB * kJ * 216) return;
  int c = idx % 216;
  int row = idx / 216;
  int b = row / kJ, j = row % kJ;
  float s = g2b[c];
  for (int e = c_nb_off[j]; e < c_nb_off[j + 1]; ++e)
    s += Wo[((size_t)b * kJ + c_nb[e]) * 224 + c];   // Wo padded to Np=224
  x2[(size_t)row * 216 + c] = s;
}

// ---------------- IDCT-II (ortho) over time ----------------
__global__ void idct_kernel(const float* __restrict__ x2, float* __restrict__ pred) {
  int idx = blockIdx.x * blockDim.x + threadIdx.x;
  if (idx >= kB * kT * kJ * kD) return;
  int d  = idx % kD;
  int j  = (idx / kD) % kJ;
  int tt = (idx / (kD * kJ)) % kT;
  int b  = idx / (kD * kJ * kT);
  const float c0 = 0.20412414523f;   // 1/sqrt(24)
  const float ck = 0.28867513459f;   // sqrt(2/24)
  const float w  = 0.06544984695f;   // pi/48
  float s = 0.0f;
  const float* src = x2 + ((size_t)b * kJ + j) * 216 + d;
#pragma unroll
  for (int k = 0; k < kT; ++k) {
    float m = (k == 0) ? c0 : ck * __cosf(w * (float)(k * (2 * tt + 1)));
    s += m * src[k * kD];
  }
  pred[idx] = s;
}

// ---------------- launch ----------------
extern "C" void kernel_launch(void* const* d_in, const int* in_sizes, int n_in,
                              void* d_out, int out_size, void* d_ws, size_t ws_size,
                              hipStream_t stream) {
  const float* dct_input = (const float*)d_in[0];
  const float* dct_hist  = (const float*)d_in[1];
  const float* qw1 = (const float*)d_in[2];  const float* qb1 = (const float*)d_in[3];
  const float* qw2 = (const float*)d_in[4];  const float* qb2 = (const float*)d_in[5];
  const float* kw1 = (const float*)d_in[6];  const float* kb1 = (const float*)d_in[7];
  const float* kw2 = (const float*)d_in[8];  const float* kb2 = (const float*)d_in[9];
  const float* vw1 = (const float*)d_in[10]; const float* vb1 = (const float*)d_in[11];
  const float* vw2 = (const float*)d_in[12]; const float* vb2 = (const float*)d_in[13];
  const float* g1w = (const float*)d_in[14]; const float* g1b = (const float*)d_in[15];
  const float* g2w = (const float*)d_in[16]; const float* g2b = (const float*)d_in[17];
  float* pred = (float*)d_out;

  char* p = (char*)d_ws;
  auto alloc = [&](size_t bytes) -> char* {
    char* r = p;
    p += (bytes + 255) & ~(size_t)255;
    return r;
  };
  __bf16* qw1p = (__bf16*)alloc((size_t)64 * 256 * 2);
  __bf16* qw2p = (__bf16*)alloc((size_t)256 * 256 * 2);
  __bf16* kw1p = (__bf16*)alloc((size_t)64 * 256 * 2);
  __bf16* kw2p = (__bf16*)alloc((size_t)256 * 256 * 2);
  __bf16* vw1p = (__bf16*)alloc((size_t)64 * 256 * 2);
  __bf16* vw2p = (__bf16*)alloc((size_t)256 * 256 * 2);
  __bf16* g1wp = (__bf16*)alloc((size_t)4608 * 256 * 2);
  __bf16* g2wp = (__bf16*)alloc((size_t)256 * 224 * 2);
  float*  Qws  = (float*)alloc((size_t)kB * 144 * 256 * 4);
  __bf16* feat = (__bf16*)alloc((size_t)kB * kJ * 4608 * 2);
  float*  Yb   = (float*)alloc((size_t)kB * kJ * 256 * 4);
  __bf16* x1b  = (__bf16*)alloc((size_t)kB * kJ * 256 * 2);
  float*  Wob  = (float*)alloc((size_t)kB * kJ * 224 * 4);
  float*  x2b  = (float*)alloc((size_t)kB * kJ * 216 * 4);

  auto packW = [&](const float* W, __bf16* out, int K, int Nc, int Kp, int Np) {
    int total = Kp * Np;
    pack_weight<<<(total + 255) / 256, 256, 0, stream>>>(W, out, K, Nc, Kp, Np);
  };
  packW(qw1, qw1p, 40, 256, 64, 256);
  packW(qw2, qw2p, 256, 256, 256, 256);
  packW(kw1, kw1p, 40, 256, 64, 256);
  packW(kw2, kw2p, 256, 256, 256, 256);
  packW(vw1, vw1p, 40, 256, 64, 256);
  packW(vw2, vw2p, 256, 256, 256, 256);
  packW(g1w, g1wp, 4608, 256, 4608, 256);
  packW(g2w, g2wp, 256, 216, 256, 224);

  qmlp_kernel<<<kB * kIT, 256, 0, stream>>>(dct_input, qw1p, qb1, qw2p, qb2, Qws);

  attn_kernel<<<kB * kIT, 256, 0, stream>>>(dct_hist, Qws,
                                            kw1p, kb1, kw2p, kb2,
                                            vw1p, vb1, vw2p, vb2, feat);

  gemm_kernel<<<(kB * kJ / 16) * 2, 256, 0, stream>>>(feat, g1wp, Yb, 4608, 256, 256);
  graph1_kernel<<<(kB * kJ * 256 + 255) / 256, 256, 0, stream>>>(Yb, g1b, x1b);
  gemm_kernel<<<(kB * kJ / 16) * 2, 256, 0, stream>>>(x1b, g2wp, Wob, 256, 216, 224);
  graph2_kernel<<<(kB * kJ * 216 + 255) / 256, 256, 0, stream>>>(Wob, g2b, x2b);
  idct_kernel<<<(kB * kT * kJ * kD + 255) / 256, 256, 0, stream>>>(x2b, pred);
}
